// QuantumBlock_67508295959011
// MI455X (gfx1250) — compile-verified, matrix operand order checked by reading, not run on hardware
//
#include <hip/hip_runtime.h>
#include <hip/hip_bf16.h>

// ---------------- types ----------------
typedef __bf16 bf16;
typedef __attribute__((ext_vector_type(16))) __bf16 v16bf;
typedef __attribute__((ext_vector_type(8)))  float   v8f;

#define DM    1024   // model dim
#define DFF   4096   // ffn dim
#define SEQ   1024
#define BATCH 4
#define NHEAD 16
#define DH    64
#define MTOK  (BATCH*SEQ)

#define TM 64
#define TN 64
#define TK 32

__device__ __forceinline__ bf16 f2bf(float f) {
  union { float f; unsigned u; } v; v.f = f;
  unsigned r = v.u + 0x7FFFu + ((v.u >> 16) & 1u);   // RNE
  union { unsigned short s; bf16 b; } o; o.s = (unsigned short)(r >> 16);
  return o.b;
}

__device__ __forceinline__ v8f wmma_bf16(v16bf a, v16bf b, v8f c) {
  return __builtin_amdgcn_wmma_f32_16x16x32_bf16(false, a, false, b, (short)0, c, false, false);
}

__device__ __forceinline__ float gelu_exact(float x) {
  return 0.5f * x * (1.0f + erff(x * 0.70710678118654752f));
}

// CDNA5 async global->LDS copy (ASYNCcnt path, no VGPR staging).
// VDST vgpr holds the workgroup-relative LDS byte offset (hardware adds LDS_BASE).
__device__ __forceinline__ void async_cp16(bf16* lds_dst, const bf16* gsrc) {
  unsigned loff = (unsigned)(size_t)lds_dst;   // low 32 bits of generic ptr = LDS offset
  asm volatile("global_load_async_to_lds_b128 %0, %1, off"
               :: "v"(loff), "v"(gsrc)
               : "memory");
}
__device__ __forceinline__ void wait_async0() {
  asm volatile("s_wait_asynccnt 0" ::: "memory");
}

// A-operand (and NT-B-operand) fragment from an LDS tile stored [row][k] with k-stride TK.
// 16-bit A 16x32 layout: lanes 0-15 -> M=lane, K={0..7,16..23}; lanes 16-31 -> K=+8.
__device__ __forceinline__ v16bf frag_rowK(const bf16* lds, int sub, int lane) {
  const int row = (sub << 4) + (lane & 15);
  const int kh  = (lane >> 4) << 3;
  v16bf f;
#pragma unroll
  for (int j = 0; j < 16; ++j) f[j] = lds[row * TK + j + (j & 8) + kh];
  return f;
}

__device__ __forceinline__ float redmax16(float v) {
#pragma unroll
  for (int d = 1; d < 16; d <<= 1) v = fmaxf(v, __shfl_xor(v, d, 32));
  return v;
}
__device__ __forceinline__ float redsum16(float v) {
#pragma unroll
  for (int d = 1; d < 16; d <<= 1) v += __shfl_xor(v, d, 32);
  return v;
}

// ---------------- elementwise kernels ----------------
__global__ void k_cvt(const float* __restrict__ s, bf16* __restrict__ d, size_t n) {
  size_t i = (size_t)blockIdx.x * blockDim.x + threadIdx.x;
  if (i < n) d[i] = f2bf(s[i]);
}
__global__ void k_scale_cvt(const float* __restrict__ s, const float* __restrict__ sp,
                            float sc, bf16* __restrict__ db, float* __restrict__ df, size_t n) {
  size_t i = (size_t)blockIdx.x * blockDim.x + threadIdx.x;
  if (i >= n) return;
  float k = sc * (sp ? sp[0] : 1.0f);
  float v = s[i] * k;
  if (db) db[i] = f2bf(v);
  if (df) df[i] = v;
}
__global__ void k_zero(float* __restrict__ p, size_t n) {
  size_t i = (size_t)blockIdx.x * blockDim.x + threadIdx.x;
  if (i < n) p[i] = 0.0f;
}
__global__ void k_eye(float* __restrict__ a, int d) {
  int i = blockIdx.x * blockDim.x + threadIdx.x;
  if (i < d) a[(size_t)i * d + i] += 1.0f;
}
__global__ void k_accum(float* __restrict__ acc, const float* __restrict__ p,
                        float sgn, bf16* __restrict__ pb, size_t n) {
  size_t i = (size_t)blockIdx.x * blockDim.x + threadIdx.x;
  if (i >= n) return;
  float v = p[i];
  acc[i] += sgn * v;
  pb[i] = f2bf(v);
}

// ---------------- complex NT GEMM: out = A @ W^T (+bias)(+gelu)(+res) ----------------
// A: [M,K] complex bf16 (r,i planes). W: [Nd,K] complex bf16. WMMA f32 accumulate.
// Double-buffered K loop: async global->LDS copies of slab k+1 overlap WMMA on slab k.
__global__ __launch_bounds__(256) void k_cgemm(
    const bf16* __restrict__ Ar, const bf16* __restrict__ Ai,
    const bf16* __restrict__ Wr, const bf16* __restrict__ Wi,
    const float* __restrict__ br, const float* __restrict__ bi,
    const float* __restrict__ resr, const float* __restrict__ resi,
    float* __restrict__ outr, float* __restrict__ outi,
    bf16* __restrict__ obr, bf16* __restrict__ obi,
    float* __restrict__ outc,
    int M, int Nd, int K, int do_gelu)
{
  __shared__ __align__(16) bf16 lAr[2][TM * TK], lAi[2][TM * TK];
  __shared__ __align__(16) bf16 lWr[2][TN * TK], lWi[2][TN * TK];
  const int tid = threadIdx.x;
  const int lane = tid & 31, wave = tid >> 5;
  const int msub = wave & 3, npair = wave >> 2;     // 4 m-subs x 2 n-pairs
  const int m0 = blockIdx.y * TM, n0 = blockIdx.x * TN;

  v8f acc_rp[2] = {{}, {}}, acc_rm[2] = {{}, {}}, acc_i[2] = {{}, {}};

  const int lrow = tid >> 2, lcol = (tid & 3) * 8;   // 256 threads cover 64x32 in 8-elem chunks
  const size_t gA = (size_t)(m0 + lrow) * K + lcol;
  const size_t gW = (size_t)(n0 + lrow) * K + lcol;
  const int lidx = lrow * TK + lcol;

  // prologue: fill buffer 0
  async_cp16(&lAr[0][lidx], &Ar[gA]);
  async_cp16(&lAi[0][lidx], &Ai[gA]);
  async_cp16(&lWr[0][lidx], &Wr[gW]);
  async_cp16(&lWi[0][lidx], &Wi[gW]);
  wait_async0();
  __syncthreads();

  int cur = 0;
  for (int k0 = 0; k0 < K; k0 += TK) {
    if (k0 + TK < K) {                      // prefetch next slab into the other buffer
      int nxt = cur ^ 1;
      async_cp16(&lAr[nxt][lidx], &Ar[gA + k0 + TK]);
      async_cp16(&lAi[nxt][lidx], &Ai[gA + k0 + TK]);
      async_cp16(&lWr[nxt][lidx], &Wr[gW + k0 + TK]);
      async_cp16(&lWi[nxt][lidx], &Wi[gW + k0 + TK]);
    }
    v16bf ar = frag_rowK(lAr[cur], msub, lane);
    v16bf ai = frag_rowK(lAi[cur], msub, lane);
#pragma unroll
    for (int t = 0; t < 2; ++t) {
      int nsub = npair * 2 + t;
      v16bf wr = frag_rowK(lWr[cur], nsub, lane);
      v16bf wi = frag_rowK(lWi[cur], nsub, lane);
      acc_rp[t] = wmma_bf16(ar, wr, acc_rp[t]);   // +Ar*Wr
      acc_rm[t] = wmma_bf16(ai, wi, acc_rm[t]);   // +Ai*Wi  (subtract in epilogue)
      acc_i [t] = wmma_bf16(ar, wi, acc_i [t]);   // +Ar*Wi
      acc_i [t] = wmma_bf16(ai, wr, acc_i [t]);   // +Ai*Wr
    }
    wait_async0();       // next slab landed in LDS
    __syncthreads();     // all waves done reading cur before it is overwritten next iter
    cur ^= 1;
  }

  const int half = lane >> 4;
#pragma unroll
  for (int t = 0; t < 2; ++t) {
    int ng = n0 + (npair * 2 + t) * 16 + (lane & 15);
    float vbr = br ? br[ng] : 0.0f;
    float vbi = bi ? bi[ng] : 0.0f;
#pragma unroll
    for (int r = 0; r < 8; ++r) {
      int mg = m0 + msub * 16 + r + half * 8;
      size_t off = (size_t)mg * Nd + ng;
      float vr = acc_rp[t][r] - acc_rm[t][r] + vbr;
      float vi = acc_i [t][r] + vbi;
      if (do_gelu) { vr = gelu_exact(vr); vi = gelu_exact(vi); }
      if (resr) { vr += resr[off]; vi += resi[off]; }
      if (outr) { outr[off] = vr; outi[off] = vi; }
      if (obr)  { obr[off] = f2bf(vr); obi[off] = f2bf(vi); }
      if (outc) { outc[off * 2] = vr; outc[off * 2 + 1] = vi; }
    }
  }
}

// ---------------- real GEMM (row-major x row-major), C = alpha * A@B ----------------
__global__ __launch_bounds__(256) void k_rgemm(
    const bf16* __restrict__ A, const bf16* __restrict__ B, float* __restrict__ C,
    float alpha, int M, int Nd, int K)
{
  __shared__ __align__(16) bf16 lA[TM * TK], lB[TK * TN];
  const int tid = threadIdx.x;
  const int lane = tid & 31, wave = tid >> 5;
  const int msub = wave & 3, npair = wave >> 2;
  const int m0 = blockIdx.y * TM, n0 = blockIdx.x * TN;

  v8f acc[2] = {{}, {}};
  const int arow = tid >> 2, acol = (tid & 3) * 8;
  const int bk = tid >> 3, bn = (tid & 7) * 8;
  for (int k0 = 0; k0 < K; k0 += TK) {
    __syncthreads();
    *(uint4*)&lA[arow * TK + acol] = *(const uint4*)&A[(size_t)(m0 + arow) * K + k0 + acol];
    *(uint4*)&lB[bk * TN + bn]     = *(const uint4*)&B[(size_t)(k0 + bk) * Nd + n0 + bn];
    __syncthreads();
    v16bf af = frag_rowK(lA, msub, lane);
#pragma unroll
    for (int t = 0; t < 2; ++t) {
      int n = (npair * 2 + t) * 16 + (lane & 15);
      int kh = (lane >> 4) << 3;
      v16bf bfr;
#pragma unroll
      for (int j = 0; j < 16; ++j) bfr[j] = lB[(j + (j & 8) + kh) * TN + n];
      acc[t] = wmma_bf16(af, bfr, acc[t]);
    }
  }
  const int half = lane >> 4;
#pragma unroll
  for (int t = 0; t < 2; ++t) {
    int ng = n0 + (npair * 2 + t) * 16 + (lane & 15);
#pragma unroll
    for (int r = 0; r < 8; ++r) {
      int mg = m0 + msub * 16 + r + half * 8;
      C[(size_t)mg * Nd + ng] = alpha * acc[t][r];
    }
  }
}

// ---------------- flash attention, complex scores, one wave per 16-query tile ----------------
__global__ __launch_bounds__(32) void k_attn(
    const bf16* __restrict__ Qr, const bf16* __restrict__ Qi,
    const bf16* __restrict__ Kr, const bf16* __restrict__ Ki,
    const bf16* __restrict__ Vr, const bf16* __restrict__ Vi,
    bf16* __restrict__ Or_, bf16* __restrict__ Oi_)
{
  const int lane = threadIdx.x & 31;
  const int qt = blockIdx.x, h = blockIdx.y, b = blockIdx.z;
  const int rowbase = b * SEQ + qt * 16;
  const int cbase = h * DH;
  const int half = lane >> 4, kh = half << 3;

  __shared__ __align__(16) bf16 lKr[32 * DH], lKi[32 * DH], lVr[32 * DH], lVi[32 * DH];
  __shared__ float lP[16 * 32];

  // Q fragments: 2 k-steps (dh=64), real + imag
  v16bf qr[2], qi[2];
  {
    int m = rowbase + (lane & 15);
#pragma unroll
    for (int ks = 0; ks < 2; ++ks)
#pragma unroll
      for (int j = 0; j < 16; ++j) {
        int d = ks * 32 + j + (j & 8) + kh;
        qr[ks][j] = Qr[(size_t)m * DM + cbase + d];
        qi[ks][j] = Qi[(size_t)m * DM + cbase + d];
      }
  }

  float rmax[8], rsum[8];
#pragma unroll
  for (int r = 0; r < 8; ++r) { rmax[r] = -3.0e38f; rsum[r] = 0.0f; }
  v8f o_r[4] = {{}, {}, {}, {}}, o_i[4] = {{}, {}, {}, {}};

  for (int j0 = 0; j0 < SEQ; j0 += 32) {
#pragma unroll
    for (int c = 0; c < 8; ++c) {
      int chunk = c * 32 + lane;                 // 256 chunks of 8 bf16
      int key = chunk >> 3, dg = (chunk & 7) * 8;
      size_t g = (size_t)(b * SEQ + j0 + key) * DM + cbase + dg;
      *(uint4*)&lKr[key * DH + dg] = *(const uint4*)&Kr[g];
      *(uint4*)&lKi[key * DH + dg] = *(const uint4*)&Ki[g];
      *(uint4*)&lVr[key * DH + dg] = *(const uint4*)&Vr[g];
      *(uint4*)&lVi[key * DH + dg] = *(const uint4*)&Vi[g];
    }
    __syncthreads();

    // complex scores for two 16-key tiles: sim = Q conj(K)^T
    float s[2][8];
#pragma unroll
    for (int nt = 0; nt < 2; ++nt) {
      v8f srp = {}, sip = {}, sim = {};
      int n = nt * 16 + (lane & 15);
#pragma unroll
      for (int ks = 0; ks < 2; ++ks) {
        v16bf kr, ki;
#pragma unroll
        for (int j = 0; j < 16; ++j) {
          int d = ks * 32 + j + (j & 8) + kh;
          kr[j] = lKr[n * DH + d];
          ki[j] = lKi[n * DH + d];
        }
        srp = wmma_bf16(qr[ks], kr, srp);   // +QrKr
        srp = wmma_bf16(qi[ks], ki, srp);   // +QiKi
        sip = wmma_bf16(qi[ks], kr, sip);   // +QiKr
        sim = wmma_bf16(qr[ks], ki, sim);   // +QrKi (subtract)
      }
#pragma unroll
      for (int r = 0; r < 8; ++r) {
        float re = srp[r], im = sip[r] - sim[r];
        s[nt][r] = sqrtf(re * re + im * im) * 0.125f;   // /sqrt(64)
      }
    }

    // online softmax per row
    float fsc[8];
#pragma unroll
    for (int r = 0; r < 8; ++r) {
      float mx = redmax16(fmaxf(s[0][r], s[1][r]));
      float mn = fmaxf(rmax[r], mx);
      float f = __expf(rmax[r] - mn);
      float p0 = __expf(s[0][r] - mn), p1 = __expf(s[1][r] - mn);
      rsum[r] = rsum[r] * f + redsum16(p0 + p1);
      rmax[r] = mn; fsc[r] = f;
      s[0][r] = p0; s[1][r] = p1;
    }
#pragma unroll
    for (int dt = 0; dt < 4; ++dt)
#pragma unroll
      for (int r = 0; r < 8; ++r) { o_r[dt][r] *= fsc[r]; o_i[dt][r] *= fsc[r]; }

    // re-layout P (C layout -> A fragment) through LDS
#pragma unroll
    for (int nt = 0; nt < 2; ++nt)
#pragma unroll
      for (int r = 0; r < 8; ++r)
        lP[(r + 8 * half) * 32 + nt * 16 + (lane & 15)] = s[nt][r];
    __syncthreads();
    v16bf pf;
    {
      int m = lane & 15;
#pragma unroll
      for (int j = 0; j < 16; ++j) pf[j] = f2bf(lP[m * 32 + j + (j & 8) + kh]);
    }

    // O += P @ V  (P real, V complex), 4 dh-tiles
#pragma unroll
    for (int dt = 0; dt < 4; ++dt) {
      int n = dt * 16 + (lane & 15);
      v16bf vr, vi;
#pragma unroll
      for (int j = 0; j < 16; ++j) {
        int key = j + (j & 8) + kh;
        vr[j] = lVr[key * DH + n];
        vi[j] = lVi[key * DH + n];
      }
      o_r[dt] = wmma_bf16(pf, vr, o_r[dt]);
      o_i[dt] = wmma_bf16(pf, vi, o_i[dt]);
    }
    __syncthreads();
  }

  float inv[8];
#pragma unroll
  for (int r = 0; r < 8; ++r) inv[r] = 1.0f / rsum[r];
#pragma unroll
  for (int dt = 0; dt < 4; ++dt)
#pragma unroll
    for (int r = 0; r < 8; ++r) {
      size_t off = (size_t)(rowbase + r + 8 * half) * DM + cbase + dt * 16 + (lane & 15);
      Or_[off] = f2bf(o_r[dt][r] * inv[r]);
      Oi_[off] = f2bf(o_i[dt][r] * inv[r]);
    }
}

// ---------------- host orchestration ----------------
static inline dim3 g1(size_t n) { return dim3((unsigned)((n + 255) / 256)); }

extern "C" void kernel_launch(void* const* d_in, const int* in_sizes, int n_in,
                              void* d_out, int out_size, void* d_ws, size_t ws_size,
                              hipStream_t stream) {
  (void)in_sizes; (void)n_in; (void)out_size; (void)ws_size;
  const float* x_r = (const float*)d_in[0];
  const float* x_i = (const float*)d_in[1];
  const float *qWr = (const float*)d_in[2],  *qWi = (const float*)d_in[3];
  const float *qbr = (const float*)d_in[4],  *qbi = (const float*)d_in[5];
  const float *kWr = (const float*)d_in[6],  *kWi = (const float*)d_in[7];
  const float *kbr = (const float*)d_in[8],  *kbi = (const float*)d_in[9];
  const float *vWr = (const float*)d_in[10], *vWi = (const float*)d_in[11];
  const float *vbr = (const float*)d_in[12], *vbi = (const float*)d_in[13];
  const float *oWr = (const float*)d_in[14], *oWi = (const float*)d_in[15];
  const float *obr_ = (const float*)d_in[16], *obi_ = (const float*)d_in[17];
  const float *f1Wr = (const float*)d_in[18], *f1Wi = (const float*)d_in[19];
  const float *f1br = (const float*)d_in[20], *f1bi = (const float*)d_in[21];
  const float *f2Wr = (const float*)d_in[22], *f2Wi = (const float*)d_in[23];
  const float *f2br = (const float*)d_in[24], *f2bi = (const float*)d_in[25];
  const float *H = (const float*)d_in[26];
  const float *dtp = (const float*)d_in[27];

  char* ws = (char*)d_ws;
  size_t off = 0;
  auto alloc = [&](size_t bytes) -> void* {
    void* p = ws + off; off += (bytes + 255) & ~(size_t)255; return p;
  };
  const size_t nTok = (size_t)MTOK * DM;        // 4M
  const size_t nFF  = (size_t)MTOK * DFF;       // 16M
  const size_t nW   = (size_t)DM * DM;          // 1M
  const size_t nWff = (size_t)DFF * DM;         // 4M

  bf16 *xbr = (bf16*)alloc(nTok * 2), *xbi = (bf16*)alloc(nTok * 2);
  bf16 *wq_r = (bf16*)alloc(nW * 2), *wq_i = (bf16*)alloc(nW * 2);
  bf16 *wk_r = (bf16*)alloc(nW * 2), *wk_i = (bf16*)alloc(nW * 2);
  bf16 *wv_r = (bf16*)alloc(nW * 2), *wv_i = (bf16*)alloc(nW * 2);
  bf16 *wo_r = (bf16*)alloc(nW * 2), *wo_i = (bf16*)alloc(nW * 2);
  bf16 *w1_r = (bf16*)alloc(nWff * 2), *w1_i = (bf16*)alloc(nWff * 2);
  bf16 *w2_r = (bf16*)alloc(nWff * 2), *w2_i = (bf16*)alloc(nWff * 2);
  bf16 *Qbr = (bf16*)alloc(nTok * 2), *Qbi = (bf16*)alloc(nTok * 2);
  bf16 *Kbr = (bf16*)alloc(nTok * 2), *Kbi = (bf16*)alloc(nTok * 2);
  bf16 *Vbr = (bf16*)alloc(nTok * 2), *Vbi = (bf16*)alloc(nTok * 2);
  bf16 *Obr = (bf16*)alloc(nTok * 2), *Obi = (bf16*)alloc(nTok * 2);
  float *x1r = (float*)alloc(nTok * 4), *x1i = (float*)alloc(nTok * 4);
  bf16 *x1br = (bf16*)alloc(nTok * 2), *x1bi = (bf16*)alloc(nTok * 2);
  bf16 *ffr = (bf16*)alloc(nFF * 2), *ffi = (bf16*)alloc(nFF * 2);
  float *x2r = (float*)alloc(nTok * 4), *x2i = (float*)alloc(nTok * 4);
  bf16 *x2br = (bf16*)alloc(nTok * 2), *x2bi = (bf16*)alloc(nTok * 2);
  float *A1f = (float*)alloc(nW * 4);
  bf16  *A1b = (bf16*)alloc(nW * 2);
  float *Cacc = (float*)alloc(nW * 4), *Sacc = (float*)alloc(nW * 4);
  float *Pf = (float*)alloc(nW * 4);
  bf16 *Pb0 = (bf16*)alloc(nW * 2), *Pb1 = (bf16*)alloc(nW * 2);
  bf16 *Ubr = (bf16*)alloc(nW * 2), *Ubi = (bf16*)alloc(nW * 2);

  // ---- convert inputs/weights to bf16 ----
  k_cvt<<<g1(nTok), 256, 0, stream>>>(x_r, xbr, nTok);
  k_cvt<<<g1(nTok), 256, 0, stream>>>(x_i, xbi, nTok);
  k_cvt<<<g1(nW), 256, 0, stream>>>(qWr, wq_r, nW);  k_cvt<<<g1(nW), 256, 0, stream>>>(qWi, wq_i, nW);
  k_cvt<<<g1(nW), 256, 0, stream>>>(kWr, wk_r, nW);  k_cvt<<<g1(nW), 256, 0, stream>>>(kWi, wk_i, nW);
  k_cvt<<<g1(nW), 256, 0, stream>>>(vWr, wv_r, nW);  k_cvt<<<g1(nW), 256, 0, stream>>>(vWi, wv_i, nW);
  k_cvt<<<g1(nW), 256, 0, stream>>>(oWr, wo_r, nW);  k_cvt<<<g1(nW), 256, 0, stream>>>(oWi, wo_i, nW);
  k_cvt<<<g1(nWff), 256, 0, stream>>>(f1Wr, w1_r, nWff); k_cvt<<<g1(nWff), 256, 0, stream>>>(f1Wi, w1_i, nWff);
  k_cvt<<<g1(nWff), 256, 0, stream>>>(f2Wr, w2_r, nWff); k_cvt<<<g1(nWff), 256, 0, stream>>>(f2Wi, w2_i, nWff);

  // ---- Q,K,V projections (bf16 out only) ----
  dim3 blk(256);
  dim3 gproj(DM / TN, MTOK / TM);
  k_cgemm<<<gproj, blk, 0, stream>>>(xbr, xbi, wq_r, wq_i, qbr, qbi, nullptr, nullptr,
                                     nullptr, nullptr, Qbr, Qbi, nullptr, MTOK, DM, DM, 0);
  k_cgemm<<<gproj, blk, 0, stream>>>(xbr, xbi, wk_r, wk_i, kbr, kbi, nullptr, nullptr,
                                     nullptr, nullptr, Kbr, Kbi, nullptr, MTOK, DM, DM, 0);
  k_cgemm<<<gproj, blk, 0, stream>>>(xbr, xbi, wv_r, wv_i, vbr, vbi, nullptr, nullptr,
                                     nullptr, nullptr, Vbr, Vbi, nullptr, MTOK, DM, DM, 0);

  // ---- attention ----
  k_attn<<<dim3(SEQ / 16, NHEAD, BATCH), dim3(32), 0, stream>>>(Qbr, Qbi, Kbr, Kbi, Vbr, Vbi, Obr, Obi);

  // ---- output proj + residual -> x1 ----
  k_cgemm<<<gproj, blk, 0, stream>>>(Obr, Obi, wo_r, wo_i, obr_, obi_, x_r, x_i,
                                     x1r, x1i, x1br, x1bi, nullptr, MTOK, DM, DM, 0);

  // ---- FFN: gelu(fc1) then fc2 + residual -> x2 ----
  k_cgemm<<<dim3(DFF / TN, MTOK / TM), blk, 0, stream>>>(x1br, x1bi, w1_r, w1_i, f1br, f1bi,
                                     nullptr, nullptr, nullptr, nullptr, ffr, ffi, nullptr,
                                     MTOK, DFF, DM, 1);
  k_cgemm<<<gproj, blk, 0, stream>>>(ffr, ffi, w2_r, w2_i, f2br, f2bi, x1r, x1i,
                                     x2r, x2i, x2br, x2bi, nullptr, MTOK, DM, DFF, 0);

  // ---- U = exp(-i H dt) = cos(A) - i sin(A), A = H*dt, Taylor to degree 6 ----
  k_scale_cvt<<<g1(nW), 256, 0, stream>>>(H, dtp, 1.0f, A1b, A1f, nW);
  k_zero<<<g1(nW), 256, 0, stream>>>(Cacc, nW);
  k_zero<<<g1(nW), 256, 0, stream>>>(Sacc, nW);
  k_eye<<<g1(DM), 256, 0, stream>>>(Cacc, DM);
  k_accum<<<g1(nW), 256, 0, stream>>>(Sacc, A1f, 1.0f, Pb0, nW);   // sin += A; Pb0 = bf16(A)

  bf16* pb[2] = { Pb0, Pb1 };
  int cur = 0;
  dim3 gU(DM / TN, DM / TM);
  for (int k = 2; k <= 6; ++k) {
    k_rgemm<<<gU, blk, 0, stream>>>(pb[cur], A1b, Pf, 1.0f / (float)k, DM, DM, DM);
    float sgn; float* tgt;
    if ((k & 1) == 0) { tgt = Cacc; sgn = ((k / 2) & 1) ? -1.0f : 1.0f; }
    else              { tgt = Sacc; sgn = (((k - 1) / 2) & 1) ? -1.0f : 1.0f; }
    k_accum<<<g1(nW), 256, 0, stream>>>(tgt, Pf, sgn, pb[cur ^ 1], nW);
    cur ^= 1;
  }
  k_cvt<<<g1(nW), 256, 0, stream>>>(Cacc, Ubr, nW);                       // Wr =  cos(A)
  k_scale_cvt<<<g1(nW), 256, 0, stream>>>(Sacc, nullptr, -1.0f, Ubi, nullptr, nW); // Wi = -sin(A)

  // ---- out = x2 @ U (U symmetric -> NT gemm works), interleaved complex to d_out ----
  k_cgemm<<<gproj, blk, 0, stream>>>(x2br, x2bi, Ubr, Ubi, nullptr, nullptr, nullptr, nullptr,
                                     nullptr, nullptr, nullptr, nullptr, (float*)d_out,
                                     MTOK, DM, DM, 0);
}
